// LSTMSimple_85375359910514
// MI455X (gfx1250) — matrix operand order repaired and checked by reading, not compiled, not run on hardware
//
#include <hip/hip_runtime.h>
#include <hip/hip_bf16.h>
#include <math.h>

// Problem constants (from reference): T=2048, B=32, I=1024, H=1024, A=512, O=1
#define T_STEPS 2048
#define BB      32
#define II      1024
#define HH      1024
#define AATT    512
#define G4H     4096   // 4*H

typedef __bf16 bf16;
typedef __attribute__((ext_vector_type(16))) __bf16 v16bf;
typedef __attribute__((ext_vector_type(8)))  __bf16 v8bf;
typedef __attribute__((ext_vector_type(8)))  float  v8f;
typedef __attribute__((ext_vector_type(4)))  unsigned v4u;
typedef __attribute__((ext_vector_type(8)))  int      v8i;
typedef __attribute__((ext_vector_type(4)))  int      v4i;

// ---------------------------------------------------------------------------
// WMMA fragment helpers (CDNA5 16x16x32 bf16 layouts, CDNA5 ISA §7.12.2)
//
// A (16x32, MxK), lane l<16 : row m0+l,    K = k0+0..7 (v0-3), k0+16..23 (v4-7)
//                 lane l>=16: row m0+l-16, K = k0+8..15,       k0+24..31
// B (32x16, KxN), lane n<16 : col n0+n,    K = k0+0..7, k0+16..23 (from [N,K])
//                 lane n>=16: col n0+n-16, K = k0+8..15, k0+24..31
// C/D (16x16)   , vgpr r, lane n<16: (m0+r, n0+n); lane n>=16: (m0+8+r, n0+n-16)
// Works for both global and LDS-resident sources (addrspace inferred -> ds_load).
// ---------------------------------------------------------------------------

__device__ __forceinline__ v16bf load_frag_a_bf(const bf16* A, int lda, int m0, int k0) {
  const int lane = threadIdx.x & 31;
  const int l  = lane & 15;
  const int kh = (lane >> 4) * 8;
  const bf16* p = A + (size_t)(m0 + l) * lda + (k0 + kh);
  v8bf lo = *reinterpret_cast<const v8bf*>(p);        // K = k0+kh+0..7
  v8bf hi = *reinterpret_cast<const v8bf*>(p + 16);   // K = k0+kh+16..23
  return __builtin_shufflevector(lo, hi, 0,1,2,3,4,5,6,7,8,9,10,11,12,13,14,15);
}

__device__ __forceinline__ v16bf load_frag_a_f32(const float* __restrict__ A,
                                                 int lda, int m0, int k0) {
  const int lane = threadIdx.x & 31;
  const int l  = lane & 15;
  const int kh = (lane >> 4) * 8;
  const float* p = A + (size_t)(m0 + l) * lda + (k0 + kh);
  v16bf r;
#pragma unroll
  for (int i = 0; i < 8; ++i) r[i] = (bf16)p[i];
#pragma unroll
  for (int i = 0; i < 8; ++i) r[8 + i] = (bf16)p[16 + i];
  return r;
}

// W stored row-major [N, K] (i.e. original torch weight layout [out, in])
__device__ __forceinline__ v16bf load_frag_b_bf(const bf16* W, int ldw, int n0, int k0) {
  const int lane = threadIdx.x & 31;
  const int n  = lane & 15;
  const int kh = (lane >> 4) * 8;
  const bf16* p = W + (size_t)(n0 + n) * ldw + (k0 + kh);
  v8bf lo = *reinterpret_cast<const v8bf*>(p);
  v8bf hi = *reinterpret_cast<const v8bf*>(p + 16);
  return __builtin_shufflevector(lo, hi, 0,1,2,3,4,5,6,7,8,9,10,11,12,13,14,15);
}

__device__ __forceinline__ v8f wmma_bf16(v16bf a, v16bf b, v8f c) {
  // (neg_a, A, neg_b, B, c_mod, C, reuse_a, reuse_b)
  return __builtin_amdgcn_wmma_f32_16x16x32_bf16(false, a, false, b,
                                                 (short)0, c, false, false);
}

__device__ __forceinline__ float sigf(float x) { return 1.0f / (1.0f + expf(-x)); }

// ---------------------------------------------------------------------------
// CDNA5 async global->LDS copy (16B per lane), tracked with ASYNCcnt.
// GV addressing: global_load_async_to_lds_b128 vdst_lds, v[addr:addr+1], off
// lds_off = LDS byte offset (low 32 bits of generic pointer to __shared__).
// ---------------------------------------------------------------------------
__device__ __forceinline__ void async_copy_b128(unsigned lds_off, const void* gptr) {
  asm volatile("global_load_async_to_lds_b128 %0, %1, off"
               :: "v"(lds_off), "v"((unsigned long long)(uintptr_t)gptr)
               : "memory");
}
__device__ __forceinline__ void wait_asynccnt0() {
  asm volatile("s_wait_asynccnt 0" ::: "memory");
}
__device__ __forceinline__ void wait_tensorcnt0() {
#if __has_builtin(__builtin_amdgcn_s_wait_tensorcnt)
  __builtin_amdgcn_s_wait_tensorcnt(0);
#else
  asm volatile("s_wait_tensorcnt 0" ::: "memory");
#endif
}

// ---------------------------------------------------------------------------
// CDNA5 Tensor Data Mover: one descriptor-driven DMA of `bytes` contiguous
// bytes from global to LDS (D# per CDNA5 ISA).
// 1-D tile in 8-byte units: tensor_dim0 = tile_dim0 = bytes/8 (<= 65535).
// Issued once per wave; tracked with TENSORcnt.
// ---------------------------------------------------------------------------
#if __has_builtin(__builtin_amdgcn_tensor_load_to_lds)
#define HAVE_TDM 1
__device__ __forceinline__ void tdm_load_to_lds(unsigned lds_off, const void* gsrc,
                                                unsigned bytes) {
  const unsigned units = bytes >> 3;                    // data_size = 8B
  const unsigned long long ga = (unsigned long long)(uintptr_t)gsrc;
  v4u g0 = { 1u,                                        // count=1 (valid user D#)
             lds_off,                                   // lds_addr (bytes)
             (unsigned)ga,                              // global_addr[31:0]
             (unsigned)((ga >> 32) & 0x01FFFFFFu) | 0x80000000u }; // ga[56:32]|type=2
  v8i g1 = { (int)(3u << 16),                           // wg_mask=0, data_size=3 (8B)
             (int)((units & 0xFFFFu) << 16),            // tensor_dim0[15:0] @bits63:48
             (int)(((units >> 16) & 0xFFFFu) | (1u << 16)), // t_dim0[31:16], t_dim1=1
             (int)((units & 0xFFFFu) << 16),            // tile_dim0 @bits127:112
             (int)1,                                    // tile_dim1=1, tile_dim2=0
             (int)units,                                // tensor_dim0_stride[31:0]
             0, 0 };                                    // stride hi / dim1_stride = 0
  v4i g2 = {0, 0, 0, 0};
  v4i g3 = {0, 0, 0, 0};
#if defined(__clang_major__) && (__clang_major__ >= 23)
  v8i gx = {0, 0, 0, 0, 0, 0, 0, 0};
  __builtin_amdgcn_tensor_load_to_lds(g0, g1, g2, g3, gx, 0);
#else
  __builtin_amdgcn_tensor_load_to_lds(g0, g1, g2, g3, 0);
#endif
}
#endif

// ---------------------------------------------------------------------------
// Kernel 0: precision conversion / layout prep
// ---------------------------------------------------------------------------
__global__ void k_convert(const float* __restrict__ Wih, const float* __restrict__ Whh,
                          const float* __restrict__ attW,
                          const float* __restrict__ b_ih, const float* __restrict__ b_hh,
                          const float* __restrict__ h0, const float* __restrict__ c0,
                          bf16* __restrict__ Wih_bf, bf16* __restrict__ Whh_bf,
                          bf16* __restrict__ attWt_bf, float* __restrict__ bias,
                          bf16* __restrict__ h_bf, float* __restrict__ c) {
  const size_t NW = (size_t)G4H * II;       // 4,194,304
  const size_t NA = (size_t)AATT * HH;      // 524,288
  const size_t NH = (size_t)BB * HH;        // 32,768
  const size_t total = 2 * NW + NA + G4H + 2 * NH;
  for (size_t i = (size_t)blockIdx.x * blockDim.x + threadIdx.x; i < total;
       i += (size_t)gridDim.x * blockDim.x) {
    if (i < NW) {
      Wih_bf[i] = (bf16)Wih[i];
    } else if (i < 2 * NW) {
      Whh_bf[i - NW] = (bf16)Whh[i - NW];
    } else if (i < 2 * NW + NA) {
      size_t j = i - 2 * NW;                 // attWt[a][h] = attW[h][a]
      size_t a = j / HH, h = j % HH;
      attWt_bf[j] = (bf16)attW[h * AATT + a];
    } else if (i < 2 * NW + NA + G4H) {
      size_t g = i - (2 * NW + NA);
      bias[g] = b_ih[g] + b_hh[g];
    } else if (i < 2 * NW + NA + G4H + NH) {
      size_t j = i - (2 * NW + NA + G4H);
      h_bf[j] = (bf16)h0[j];
    } else {
      size_t j = i - (2 * NW + NA + G4H + NH);
      c[j] = c0[j];
    }
  }
}

// ---------------------------------------------------------------------------
// Kernel 1: x_proj = inputs @ W_ih^T + bias   -> bf16 [T*B, 4H]
// M=65536, N=4096, K=1024. One wave computes a 32x64 tile (2x4 WMMA tiles).
// ---------------------------------------------------------------------------
__global__ void k_xproj(const float* __restrict__ inputs, const bf16* __restrict__ Wih_bf,
                        const float* __restrict__ bias, bf16* __restrict__ xproj) {
  const int wave = (int)((blockIdx.x * blockDim.x + threadIdx.x) >> 5);
  const int tilesN = G4H / 64;               // 64
  const int m0 = (wave / tilesN) * 32;
  const int n0 = (wave % tilesN) * 64;

  v8f acc[2][4];
#pragma unroll
  for (int i = 0; i < 2; ++i)
#pragma unroll
    for (int j = 0; j < 4; ++j) acc[i][j] = (v8f){0, 0, 0, 0, 0, 0, 0, 0};

  for (int k0 = 0; k0 < II; k0 += 32) {
    v16bf a0 = load_frag_a_f32(inputs, II, m0, k0);
    v16bf a1 = load_frag_a_f32(inputs, II, m0 + 16, k0);
#pragma unroll
    for (int j = 0; j < 4; ++j) {
      v16bf b = load_frag_b_bf(Wih_bf, II, n0 + 16 * j, k0);
      acc[0][j] = wmma_bf16(a0, b, acc[0][j]);
      acc[1][j] = wmma_bf16(a1, b, acc[1][j]);
    }
  }

  const int lane = threadIdx.x & 31;
  const int nlo = lane & 15, mhi = (lane >> 4) * 8;
#pragma unroll
  for (int j = 0; j < 4; ++j) {
    const int n = n0 + 16 * j + nlo;
    const float bsv = bias[n];
#pragma unroll
    for (int r = 0; r < 8; ++r) {
      xproj[(size_t)(m0 + mhi + r) * G4H + n]      = (bf16)(acc[0][j][r] + bsv);
      xproj[(size_t)(m0 + 16 + mhi + r) * G4H + n] = (bf16)(acc[1][j][r] + bsv);
    }
  }
}

// ---------------------------------------------------------------------------
// Grid-wide barrier (monotonic counter). 32 resident blocks. The trailing
// __threadfence() lowers to a device-scope acquire (global_inv) so freshly
// written h/gates are re-read past L0/WGP$.
// ---------------------------------------------------------------------------
__device__ __forceinline__ void grid_sync(unsigned* cnt, int nblk, unsigned phase) {
  __threadfence();
  __syncthreads();
  if (threadIdx.x == 0) {
    atomicAdd(cnt, 1u);
    const unsigned target = phase * (unsigned)nblk;
    while (atomicAdd(cnt, 0u) < target) __builtin_amdgcn_s_sleep(1);
  }
  __syncthreads();
  __threadfence();
}

// ---------------------------------------------------------------------------
// Kernel 2: persistent LSTM scan. 32 blocks x 256 threads = 256 waves.
//
// CDNA5 big-LDS plan (320 KB/WGP):
//   lds_whh : this WG's 128 rows of W_hh (8 waves x 16-wide N strips) = 256 KB,
//             staged ONCE via the Tensor Data Mover (TENSORcnt), resident for
//             all 2048 steps.
//   lds_h   : h_t (32x1024 bf16) = 64 KB, restaged each step via per-lane
//             async global->LDS B128 copies (ASYNCcnt).
// Inner loop is then pure ds_load_b128 + v_wmma_f32_16x16x32_bf16.
// ---------------------------------------------------------------------------
__global__ void __launch_bounds__(256) k_lstm(
    const bf16* __restrict__ xproj, const bf16* __restrict__ Whh_bf,
    bf16* __restrict__ h_bf, float* __restrict__ c,
    float* __restrict__ gates, bf16* __restrict__ outs,
    unsigned* __restrict__ barrier_cnt) {
  __shared__ bf16 lds_whh[8 * 16 * HH];   // 256 KB
  __shared__ bf16 lds_h[BB * HH];         // 64 KB  (total 320 KB = WGP max)

  const int nblk = (int)gridDim.x;
  const int wave = (int)(threadIdx.x >> 5);                 // 0..7
  const int gwave = (int)blockIdx.x * 8 + wave;             // 0..255
  const int n0 = gwave * 16;                                // global N strip
  const int lane = threadIdx.x & 31;
  const int nlo = lane & 15, mhi = (lane >> 4) * 8;
  const int tid = (int)(blockIdx.x * blockDim.x + threadIdx.x);  // 0..8191

  const unsigned whh_lds_base = (unsigned)(uintptr_t)(&lds_whh[0]);
  const unsigned h_lds_base   = (unsigned)(uintptr_t)(&lds_h[0]);

  // One-time stage of this WG's 128 W_hh rows (256 KB).
  {
    const char* src = (const char*)(Whh_bf + (size_t)blockIdx.x * 128 * HH);
#ifdef HAVE_TDM
    if (wave == 0) {                 // one TDM DMA for the whole 256 KB tile
      tdm_load_to_lds(whh_lds_base, src, 128u * HH * 2u);
      wait_tensorcnt0();
    }
#else
    for (int i = (int)threadIdx.x; i < 16384; i += 256)
      async_copy_b128(whh_lds_base + (unsigned)i * 16, src + (size_t)i * 16);
    wait_asynccnt0();
#endif
    __syncthreads();
  }

  unsigned phase = 0;
  for (int t = 0; t < T_STEPS; ++t) {
    // Restage h_t (64 KB): 4096 x 16B chunks, async global->LDS.
    for (int i = (int)threadIdx.x; i < 4096; i += 256)
      async_copy_b128(h_lds_base + (unsigned)i * 16, (const char*)h_bf + (size_t)i * 16);
    wait_asynccnt0();
    __syncthreads();

    const bf16* xp = xproj + (size_t)t * BB * G4H;
    if (t + 1 < T_STEPS)  // warm L2 for next step's x_proj strip
      __builtin_prefetch(xproj + (size_t)(t + 1) * BB * G4H + n0, 0, 1);

    v8f acc0 = (v8f){0, 0, 0, 0, 0, 0, 0, 0};
    v8f acc1 = (v8f){0, 0, 0, 0, 0, 0, 0, 0};
    const bf16* wstrip = &lds_whh[(size_t)wave * 16 * HH];  // rows n0..n0+15
#pragma unroll 4
    for (int k0 = 0; k0 < HH; k0 += 32) {
      v16bf a0 = load_frag_a_bf(lds_h, HH, 0, k0);
      v16bf a1 = load_frag_a_bf(lds_h, HH, 16, k0);
      v16bf b  = load_frag_b_bf(wstrip, HH, 0, k0);
      acc0 = wmma_bf16(a0, b, acc0);
      acc1 = wmma_bf16(a1, b, acc1);
    }

    const int n = n0 + nlo;
#pragma unroll
    for (int r = 0; r < 8; ++r) {
      const int m = mhi + r;
      gates[(size_t)m * G4H + n]        = acc0[r] + (float)xp[(size_t)m * G4H + n];
      gates[(size_t)(m + 16) * G4H + n] = acc1[r] + (float)xp[(size_t)(m + 16) * G4H + n];
    }

    grid_sync(barrier_cnt, nblk, ++phase);

    // pointwise cell update: 32*1024 = 32768 elements over 8192 threads
#pragma unroll
    for (int q = 0; q < 4; ++q) {
      const int idx = tid + q * 8192;
      const int b = idx >> 10, j = idx & 1023;
      const float ig = gates[(size_t)b * G4H + j];
      const float fg = gates[(size_t)b * G4H + 1024 + j];
      const float gg = gates[(size_t)b * G4H + 2048 + j];
      const float og = gates[(size_t)b * G4H + 3072 + j];
      const float cn = sigf(fg) * c[idx] + sigf(ig) * tanhf(gg);
      const float hn = sigf(og) * tanhf(cn);
      c[idx] = cn;
      const bf16 hb = (bf16)hn;
      h_bf[idx] = hb;
      outs[(size_t)t * (BB * HH) + idx] = hb;
    }

    grid_sync(barrier_cnt, nblk, ++phase);
  }
}

// ---------------------------------------------------------------------------
// Kernel 3: attention scores. For each row m=(t,b):
//   ait_raw[m] = sum_a tanh( (outs[m,:] @ attW)[a] + attb[a] ) * attu[a]
// One wave per 16-row M tile; WMMA over N=512, K=1024; in-register row reduce.
// ---------------------------------------------------------------------------
__global__ void k_att_score(const bf16* __restrict__ outs, const bf16* __restrict__ attWt,
                            const float* __restrict__ attb, const float* __restrict__ attu,
                            float* __restrict__ ait_raw) {
  const int wave = (int)((blockIdx.x * blockDim.x + threadIdx.x) >> 5);  // 0..4095
  const int m0 = wave * 16;
  const int lane = threadIdx.x & 31;
  const int nlo = lane & 15, mhi = (lane >> 4) * 8;

  float part[8] = {0, 0, 0, 0, 0, 0, 0, 0};
  for (int nt = 0; nt < AATT / 16; ++nt) {
    v8f acc = (v8f){0, 0, 0, 0, 0, 0, 0, 0};
    for (int k0 = 0; k0 < HH; k0 += 32) {
      v16bf a = load_frag_a_bf(outs, HH, m0, k0);
      v16bf b = load_frag_b_bf(attWt, HH, nt * 16, k0);
      acc = wmma_bf16(a, b, acc);
    }
    const int n = nt * 16 + nlo;
    const float bv = attb[n], uv = attu[n];
#pragma unroll
    for (int r = 0; r < 8; ++r) part[r] += tanhf(acc[r] + bv) * uv;
  }
  // reduce across each 16-lane half (masks 1,2,4,8 stay within the half)
#pragma unroll
  for (int r = 0; r < 8; ++r) {
    float v = part[r];
    v += __shfl_xor(v, 1); v += __shfl_xor(v, 2);
    v += __shfl_xor(v, 4); v += __shfl_xor(v, 8);
    part[r] = v;
  }
  if (nlo == 0) {
#pragma unroll
    for (int r = 0; r < 8; ++r) ait_raw[m0 + mhi + r] = part[r];
  }
}

// ---------------------------------------------------------------------------
// Kernel 4: softmax over T per batch column (32 blocks)
// ---------------------------------------------------------------------------
__global__ void k_softmax(const float* __restrict__ ait_raw, float* __restrict__ ait) {
  const int b = blockIdx.x;
  __shared__ float red[256];
  float vals[T_STEPS / 256];
  float lmax = -INFINITY;
#pragma unroll
  for (int i = 0; i < T_STEPS / 256; ++i) {
    vals[i] = ait_raw[(size_t)(threadIdx.x + i * 256) * BB + b];
    lmax = fmaxf(lmax, vals[i]);
  }
  red[threadIdx.x] = lmax; __syncthreads();
  for (int s = 128; s > 0; s >>= 1) {
    if ((int)threadIdx.x < s) red[threadIdx.x] = fmaxf(red[threadIdx.x], red[threadIdx.x + s]);
    __syncthreads();
  }
  const float m = red[0]; __syncthreads();
  float lsum = 0.f;
#pragma unroll
  for (int i = 0; i < T_STEPS / 256; ++i) { vals[i] = expf(vals[i] - m); lsum += vals[i]; }
  red[threadIdx.x] = lsum; __syncthreads();
  for (int s = 128; s > 0; s >>= 1) {
    if ((int)threadIdx.x < s) red[threadIdx.x] += red[threadIdx.x + s];
    __syncthreads();
  }
  const float inv = 1.0f / red[0];
#pragma unroll
  for (int i = 0; i < T_STEPS / 256; ++i)
    ait[(size_t)(threadIdx.x + i * 256) * BB + b] = vals[i] * inv;
}

// ---------------------------------------------------------------------------
// Kernel 5: s[t,b] = outs[t,b,:] . linW   (final linear reordered before pool)
// ---------------------------------------------------------------------------
__global__ void k_rowdot(const bf16* __restrict__ outs, const float* __restrict__ linW,
                         float* __restrict__ s) {
  const int row = (int)((blockIdx.x * blockDim.x + threadIdx.x) >> 5);  // 0..65535
  const int lane = threadIdx.x & 31;
  const bf16* p = outs + (size_t)row * HH;
  float acc = 0.f;
  for (int i = lane; i < HH; i += 32) acc += (float)p[i] * linW[i];
  for (int m = 16; m >= 1; m >>= 1) acc += __shfl_xor(acc, m);
  if (lane == 0) s[row] = acc;
}

// ---------------------------------------------------------------------------
// Kernel 6: out[b] = sum_t ait[t,b]*s[t,b] + linb
// ---------------------------------------------------------------------------
__global__ void k_final(const float* __restrict__ s, const float* __restrict__ ait,
                        const float* __restrict__ linb, float* __restrict__ out) {
  const int b = blockIdx.x;
  __shared__ float red[256];
  float acc = 0.f;
  for (int t = threadIdx.x; t < T_STEPS; t += 256)
    acc += s[(size_t)t * BB + b] * ait[(size_t)t * BB + b];
  red[threadIdx.x] = acc; __syncthreads();
  for (int st = 128; st > 0; st >>= 1) {
    if ((int)threadIdx.x < st) red[threadIdx.x] += red[threadIdx.x + st];
    __syncthreads();
  }
  if (threadIdx.x == 0) out[b] = red[0] + linb[0];
}

// ---------------------------------------------------------------------------
extern "C" void kernel_launch(void* const* d_in, const int* in_sizes, int n_in,
                              void* d_out, int out_size, void* d_ws, size_t ws_size,
                              hipStream_t stream) {
  const float* inputs = (const float*)d_in[0];   // [T,B,I]
  const float* W_ih   = (const float*)d_in[1];   // [4H,I]
  const float* W_hh   = (const float*)d_in[2];   // [4H,H]
  const float* b_ih   = (const float*)d_in[3];   // [4H]
  const float* b_hh   = (const float*)d_in[4];   // [4H]
  const float* h0     = (const float*)d_in[5];   // [B,H]
  const float* c0     = (const float*)d_in[6];   // [B,H]
  const float* attW   = (const float*)d_in[7];   // [H,A] -> bf16 [A,H] in ws
  const float* attb   = (const float*)d_in[8];   // [A]
  const float* attu   = (const float*)d_in[9];   // [A,1]
  const float* linW   = (const float*)d_in[10];  // [O,H] = [1,H]
  const float* linb   = (const float*)d_in[11];  // [O]
  float* out = (float*)d_out;                    // [B,O] = 32 floats

  // ---- workspace carve-up (256B aligned) ----
  char* ws = (char*)d_ws;
  size_t off = 0;
  auto carve = [&](size_t bytes) {
    size_t o = off;
    off = (off + bytes + 255) & ~(size_t)255;
    return o;
  };
  size_t o_cnt    = carve(256);                                  // barrier counter
  size_t o_wih    = carve((size_t)G4H * II * sizeof(bf16));      // 8 MB
  size_t o_whh    = carve((size_t)G4H * HH * sizeof(bf16));      // 8 MB
  size_t o_attwt  = carve((size_t)AATT * HH * sizeof(bf16));     // 1 MB
  size_t o_bias   = carve((size_t)G4H * sizeof(float));
  size_t o_hbf    = carve((size_t)BB * HH * sizeof(bf16));
  size_t o_c      = carve((size_t)BB * HH * sizeof(float));
  size_t o_gates  = carve((size_t)BB * G4H * sizeof(float));
  size_t o_araw   = carve((size_t)T_STEPS * BB * sizeof(float));
  size_t o_ait    = carve((size_t)T_STEPS * BB * sizeof(float));
  size_t o_s      = carve((size_t)T_STEPS * BB * sizeof(float));
  size_t o_outs   = carve((size_t)T_STEPS * BB * HH * sizeof(bf16));   // 128 MB
  size_t o_xproj  = carve((size_t)T_STEPS * BB * G4H * sizeof(bf16));  // 512 MB
  (void)ws_size; (void)in_sizes; (void)n_in; (void)out_size;

  bf16*  Wih_bf  = (bf16*)(ws + o_wih);
  bf16*  Whh_bf  = (bf16*)(ws + o_whh);
  bf16*  attWt   = (bf16*)(ws + o_attwt);
  float* bias    = (float*)(ws + o_bias);
  bf16*  h_bf    = (bf16*)(ws + o_hbf);
  float* c_buf   = (float*)(ws + o_c);
  float* gates   = (float*)(ws + o_gates);
  float* araw    = (float*)(ws + o_araw);
  float* ait     = (float*)(ws + o_ait);
  float* s_buf   = (float*)(ws + o_s);
  bf16*  outs    = (bf16*)(ws + o_outs);
  bf16*  xproj   = (bf16*)(ws + o_xproj);
  unsigned* cnt  = (unsigned*)(ws + o_cnt);

  // zero the barrier counter each launch (graph-capture-safe memset node)
  hipMemsetAsync(cnt, 0, 256, stream);

  // 0) convert weights/state to bf16 layouts
  k_convert<<<4096, 256, 0, stream>>>(W_ih, W_hh, attW, b_ih, b_hh, h0, c0,
                                      Wih_bf, Whh_bf, attWt, bias, h_bf, c_buf);

  // 1) x_proj GEMM: 131072 wave-tiles of 32x64 -> 16384 blocks x 8 waves
  k_xproj<<<16384, 256, 0, stream>>>(inputs, Wih_bf, bias, xproj);

  // 2) persistent LSTM scan: 32 blocks x 256 threads (grid-barrier resident)
  k_lstm<<<32, 256, 0, stream>>>(xproj, Whh_bf, h_bf, c_buf, gates, outs, cnt);

  // 3) attention scores: 4096 M-tiles -> 512 blocks x 8 waves
  k_att_score<<<512, 256, 0, stream>>>(outs, attWt, attb, attu, araw);

  // 4) softmax over T per batch column
  k_softmax<<<32, 256, 0, stream>>>(araw, ait);

  // 5) per-row dot with linW, then 6) weighted sum over T
  k_rowdot<<<8192, 256, 0, stream>>>(outs, linW, s_buf);
  k_final<<<32, 256, 0, stream>>>(s_buf, ait, linb, out);
}
